// MIMOOFDMModel_20315195310238
// MI455X (gfx1250) — matrix operand order repaired and checked by reading, not compiled, not run on hardware
//
#include <hip/hip_runtime.h>

// MI455X (gfx1250) fused OFDM/MIMO receiver.
// 512-pt FFT = radix-2 -> tw32 -> DFT16 (WMMA) -> tw512 -> DFT16 (WMMA).
// Complex 16x16 DFT product = 2x v_wmma_f32_16x16x32_f16 with K packed as [Re|Im].
// All trig from one shared 512-entry cos/sin table; WMMA B operands pre-packed
// in f16 LDS rows so each operand is two contiguous 16-byte LDS loads.

typedef __attribute__((ext_vector_type(16))) _Float16 v16h;
typedef __attribute__((ext_vector_type(8)))  float    v8f;

#define N_SC  512
#define CP_L  7
#define ROW_L (N_SC + CP_L)   // 519

// ---- LDS pool layout (52 KB) ----
//  [    0 .. 16383]  sSpec  float [2][2][2][512]   spectra (persistent)
//  [16384 .. 20479]  sTw    float [2][512]         cos/sin table (persistent)
//  [20480 .. 36863]  scrA   16KB  multi-use: sX (ph0-1) -> sB2 (ph3-4) -> sP (ph5)
//  [36864 .. 53247]  sB1    f16   [4][2][2][32][16] packed stage-1 B operands
#define SMEM_BYTES 53248

#define SPEC(s,r,c,k)   sSpec[((((s)*2+(r))*2+(c))*N_SC)+(k)]
#define XR(w,c,n)       sX[(((w)*2+(c))*N_SC)+(n)]
#define B1(w,q,r,l,e)   sB1[(((((w)*2+(q))*2+(r))*32+(l))*16)+(e)]
#define B2(w,q,cb,l,e)  sB2[(((((w)*2+(q))*2+(cb))*32+(l))*16)+(e)]
#define PP(t,r,c,j)     sP[((((t)*2+(r))*2+(c))*256)+(j)]

__global__ __launch_bounds__(128) void mimo_ofdm_kernel(
    const float* __restrict__ y_data, const float* __restrict__ y_ls,
    const int*   __restrict__ x_idx,
    float* __restrict__ out_x, float* __restrict__ out_b)
{
  __shared__ __align__(32) char smem[SMEM_BYTES];
  float*    sSpec = (float*)smem;
  float*    sTw   = (float*)(smem + 16384);
  char*     scrA  = smem + 20480;
  _Float16* sB1   = (_Float16*)(smem + 36864);

  float*    sX  = (float*)scrA;       // phases 0-1
  _Float16* sB2 = (_Float16*)scrA;    // phases 3-4 (per-wave windows match sX's)
  float*    sP  = (float*)scrA;       // phase 5 pilot estimates

  const int b    = blockIdx.x;
  const int tid  = threadIdx.x;
  const int wave = tid >> 5;
  const int lane = tid & 31;
  const int half = lane >> 4;   // lane half for WMMA 16-bit layouts
  const int lcol = lane & 15;

  const float TWO_PI = 6.28318530717958647692f;

  // ---- phase -1: fill cos/sin table (4 sincos per thread, once per block) ----
  #pragma unroll
  for (int i = 0; i < 4; ++i) {
    int idx = tid + 128 * i;
    float s, c;
    __sincosf(TWO_PI * (float)idx * (1.0f / 512.0f), &s, &c);
    sTw[idx]       = c;
    sTw[512 + idx] = s;
  }

  // ---- phase 0: each wave loads one (src, rx) row, strips CP ----
  const int   rx   = wave & 1;
  const int   isrc = (wave < 2) ? 0 : 1;
  const float* src = (wave < 2) ? y_data : y_ls;
  const float* rowp = src + ((size_t)(b * 2 + rx) * ROW_L + CP_L) * 2;
  #pragma unroll
  for (int i = 0; i < 16; ++i) {
    int n = lane + 32 * i;
    float2 v = *(const float2*)(rowp + 2 * n);
    XR(wave, 0, n) = v.x;
    XR(wave, 1, n) = v.y;
  }
  __syncthreads();   // sTw visible to all; sX is wave-private

  // ---- constant A matrix: DFT16 packed [A_re | A_im] along K, f16, from table ----
  // A_re[m][p] = cos(2pi m p/16), A_im[m][p] = -sin(2pi m p/16)
  // 16-bit A 16x32 layout (contiguous, matches packed B): K = e + 16*half
  v16h amat;
  #pragma unroll
  for (int e = 0; e < 16; ++e) {
    int K = e + 16 * half;
    int m = lcol;
    if (K < 16) amat[e] = (_Float16)sTw[((m * K) & 15) * 32];
    else        amat[e] = (_Float16)(-sTw[512 + ((m * (K - 16)) & 15) * 32]);
  }

  // ---- phase 1: radix-2 + W32^{p*r}; store straight into packed B1 layout ----
  // B_re = [Tre ; -Tim], B_im = [Tim ; Tre]; B row = lane, elem e = K - 16*half
  #pragma unroll
  for (int i = 0; i < 8; ++i) {
    int idx = lane + 32 * i;            // 256 (n1,p) pairs
    int n1 = idx & 15, p = idx >> 4;
    float ar = XR(wave, 0, n1 + 16 * p),       ai = XR(wave, 1, n1 + 16 * p);
    float br = XR(wave, 0, n1 + 16 * p + 256), bi = XR(wave, 1, n1 + 16 * p + 256);
    float t0r = ar + br, t0i = ai + bi;                  // r = 0 (no twiddle)
    B1(wave, 0, 0, n1,      p) = (_Float16)t0r;
    B1(wave, 0, 0, 16 + n1, p) = (_Float16)(-t0i);
    B1(wave, 1, 0, n1,      p) = (_Float16)t0i;
    B1(wave, 1, 0, 16 + n1, p) = (_Float16)t0r;
    float dr = ar - br, di = ai - bi;                    // r = 1, * W32^p
    float c = sTw[p * 16], s = sTw[512 + p * 16];
    float t1r = dr * c + di * s, t1i = di * c - dr * s;  // * exp(-i*th)
    B1(wave, 0, 1, n1,      p) = (_Float16)t1r;
    B1(wave, 0, 1, 16 + n1, p) = (_Float16)(-t1i);
    B1(wave, 1, 1, n1,      p) = (_Float16)t1i;
    B1(wave, 1, 1, 16 + n1, p) = (_Float16)t1r;
  }
  __syncthreads();   // also fences scrA reuse (sX -> sB2)

  // ---- phase 2/3: DFT16 over p (2 blocks by r) + W512^{n1*k2}; pack into B2 ----
  #pragma unroll
  for (int rblk = 0; rblk < 2; ++rblk) {
    v16h bre = *(const v16h*)&B1(wave, 0, rblk, lane, 0);
    v16h bim = *(const v16h*)&B1(wave, 1, rblk, lane, 0);
    v8f zacc = {};
    v8f cre = __builtin_amdgcn_wmma_f32_16x16x32_f16(false, amat, false, bre,
                                                     (short)0, zacc, false, false);
    v8f cim = __builtin_amdgcn_wmma_f32_16x16x32_f16(false, amat, false, bim,
                                                     (short)0, zacc, false, false);
    // C elem v -> M = v + 8*half (=s), N = lcol (=n1); k2 = rblk + 2*s
    #pragma unroll
    for (int v = 0; v < 8; ++v) {
      int s2 = v + 8 * half;
      int k2 = rblk + 2 * s2;
      int n1 = lcol;
      int idx = (n1 * k2) & 511;
      float c = sTw[idx], s = sTw[512 + idx];
      float zr = cre[v] * c + cim[v] * s;   // * exp(-i*th)
      float zi = cim[v] * c - cre[v] * s;
      int cb = k2 >> 4, n = k2 & 15;
      B2(wave, 0, cb, n,      n1) = (_Float16)zr;
      B2(wave, 0, cb, 16 + n, n1) = (_Float16)(-zi);
      B2(wave, 1, cb, n,      n1) = (_Float16)zi;
      B2(wave, 1, cb, 16 + n, n1) = (_Float16)zr;
    }
  }

  // ---- phase 4: DFT16 over n1 (2 blocks of k2), scale, store spectrum ----
  const float SQRT_PI_P = 0.2262741699796952f;            // sqrt(0.0512)
  const float scale = (wave < 2) ? (1.0f / 512.0f)
                                 : (1.0f / (512.0f * SQRT_PI_P));
  #pragma unroll
  for (int cb = 0; cb < 2; ++cb) {
    v16h bre = *(const v16h*)&B2(wave, 0, cb, lane, 0);
    v16h bim = *(const v16h*)&B2(wave, 1, cb, lane, 0);
    v8f zacc = {};
    v8f cre = __builtin_amdgcn_wmma_f32_16x16x32_f16(false, amat, false, bre,
                                                     (short)0, zacc, false, false);
    v8f cim = __builtin_amdgcn_wmma_f32_16x16x32_f16(false, amat, false, bim,
                                                     (short)0, zacc, false, false);
    #pragma unroll
    for (int v = 0; v < 8; ++v) {
      int k1 = v + 8 * half;
      int k  = 16 * cb + lcol + 32 * k1;
      SPEC(isrc, rx, 0, k) = cre[v] * scale;
      SPEC(isrc, rx, 1, k) = cim[v] * scale;
    }
  }
  __syncthreads();   // sSpec cross-wave; fences scrA reuse (sB2 -> sP)

  // ---- phase 5a: compute each pilot LS estimate exactly once into sP ----
  const float LEVV[4] = {-0.9486832980505138f, -0.31622776601683794f,
                          0.31622776601683794f, 0.9486832980505138f};
  #pragma unroll
  for (int it = 0; it < 8; ++it) {
    int idx = tid + 128 * it;        // 1024 pilots: [t][rr][j]
    int t  = idx >> 9;
    int rr = (idx >> 8) & 1;
    int j  = idx & 255;
    int c  = 2 * j + t;
    int ci = x_idx[((size_t)b * 2 + t) * N_SC + c];
    float xr = LEVV[ci & 3], xi = LEVV[(ci >> 2) & 3];
    float inv = 1.0f / (xr * xr + xi * xi);
    float lr = SPEC(1, rr, 0, c), li = SPEC(1, rr, 1, c);
    PP(t, rr, 0, j) = (lr * xr + li * xi) * inv;   // Y_l / x
    PP(t, rr, 1, j) = (li * xr - lr * xi) * inv;
  }
  __syncthreads();

  // ---- phase 5b: interpolate H, 2x2 solve, demap; 4 subcarriers / thread ----
  #pragma unroll
  for (int j4 = 0; j4 < 4; ++j4) {
    int k = tid + 128 * j4;
    float H0r[2], H0i[2], H1r[2], H1i[2];
    #pragma unroll
    for (int rr = 0; rr < 2; ++rr) {
      // tx0: pilots on even subcarriers (comb index j = c/2)
      if (k == 511)          { H0r[rr] = PP(0, rr, 0, 255); H0i[rr] = PP(0, rr, 1, 255); }
      else if ((k & 1) == 0) { int j = k >> 1;
                               H0r[rr] = PP(0, rr, 0, j);   H0i[rr] = PP(0, rr, 1, j); }
      else                   { int j = (k - 1) >> 1;
                               H0r[rr] = 0.5f * (PP(0, rr, 0, j) + PP(0, rr, 0, j + 1));
                               H0i[rr] = 0.5f * (PP(0, rr, 1, j) + PP(0, rr, 1, j + 1)); }
      // tx1: pilots on odd subcarriers (comb index j = (c-1)/2)
      if (k == 0)            { H1r[rr] = PP(1, rr, 0, 0);   H1i[rr] = PP(1, rr, 1, 0); }
      else if (k & 1)        { int j = (k - 1) >> 1;
                               H1r[rr] = PP(1, rr, 0, j);   H1i[rr] = PP(1, rr, 1, j); }
      else                   { int j = (k >> 1) - 1;
                               H1r[rr] = 0.5f * (PP(1, rr, 0, j) + PP(1, rr, 0, j + 1));
                               H1i[rr] = 0.5f * (PP(1, rr, 1, j) + PP(1, rr, 1, j + 1)); }
    }

    float y0r = SPEC(0, 0, 0, k), y0i = SPEC(0, 0, 1, k);
    float y1r = SPEC(0, 1, 0, k), y1i = SPEC(0, 1, 1, k);
    float h00r = H0r[0], h00i = H0i[0], h01r = H1r[0], h01i = H1i[0];
    float h10r = H0r[1], h10i = H0i[1], h11r = H1r[1], h11i = H1i[1];

    float detr = (h00r * h11r - h00i * h11i) - (h01r * h10r - h01i * h10i);
    float deti = (h00r * h11i + h00i * h11r) - (h01r * h10i + h01i * h10r);
    float dn   = detr * detr + deti * deti;
    float idr  =  detr / dn, idi = -deti / dn;

    float n0r = (h11r * y0r - h11i * y0i) - (h01r * y1r - h01i * y1i);
    float n0i = (h11r * y0i + h11i * y0r) - (h01r * y1i + h01i * y1r);
    float n1r = (h00r * y1r - h00i * y1i) - (h10r * y0r - h10i * y0i);
    float n1i = (h00r * y1i + h00i * y1r) - (h10r * y0i + h10i * y0r);

    const float ISP = 4.419417382415922f;   // 1/sqrt(0.0512)
    float X0r = (n0r * idr - n0i * idi) * ISP, X0i = (n0r * idi + n0i * idr) * ISP;
    float X1r = (n1r * idr - n1i * idi) * ISP, X1i = (n1r * idi + n1i * idr) * ISP;

    *(float4*)(out_x + ((size_t)b * N_SC + k) * 4) = make_float4(X0r, X0i, X1r, X1i);

    auto qax = [](float xv) -> int {
      const float T2 = 0.6324555320336759f;  // 2/sqrt(10)
      return (xv < -T2) ? 0 : (xv < 0.0f) ? 1 : (xv < T2) ? 2 : 3;
    };
    int v0 = qax(X0r) + 4 * qax(X0i);
    int v1 = qax(X1r) + 4 * qax(X1i);
    *(float4*)(out_b + ((size_t)b * 2 + 0) * (N_SC * 4) + 4 * k) =
        make_float4((float)(v0 & 1), (float)((v0 >> 1) & 1),
                    (float)((v0 >> 2) & 1), (float)((v0 >> 3) & 1));
    *(float4*)(out_b + ((size_t)b * 2 + 1) * (N_SC * 4) + 4 * k) =
        make_float4((float)(v1 & 1), (float)((v1 >> 1) & 1),
                    (float)((v1 >> 2) & 1), (float)((v1 >> 3) & 1));
  }
}

extern "C" void kernel_launch(void* const* d_in, const int* in_sizes, int n_in,
                              void* d_out, int out_size, void* d_ws, size_t ws_size,
                              hipStream_t stream) {
  const float* y_data = (const float*)d_in[0];
  const float* y_ls   = (const float*)d_in[1];
  const int*   x_idx  = (const int*)d_in[2];
  const int B = in_sizes[0] / (2 * ROW_L * 2);   // [B, 2, 519, 2]
  float* out_x = (float*)d_out;
  float* out_b = out_x + (size_t)B * N_SC * 2 * 2;  // x_hat_ri then b_hat
  mimo_ofdm_kernel<<<B, 128, 0, stream>>>(y_data, y_ls, x_idx, out_x, out_b);
}
